// MultiHeadAttention_44650480009548
// MI455X (gfx1250) — compile-verified, matrix operand order checked by reading, not run on hardware
//
#include <hip/hip_runtime.h>
#include <hip/hip_bf16.h>

// MI455X (gfx1250) multi-head attention forward.
// All matmuls via v_wmma_f32_16x16x32_f16 (f16 inputs, f32 accumulate).
// attn [B,H,S,S] fp32 is written to d_out exactly once via a fused
// scores->softmax->P*V kernel that keeps the 16xS score stripe in LDS.

typedef __attribute__((ext_vector_type(16))) _Float16 v16h;
typedef __attribute__((ext_vector_type(8)))  _Float16 v8h;
typedef __attribute__((ext_vector_type(8)))  float    v8f;
typedef __attribute__((ext_vector_type(4)))  int      v4i;

typedef __attribute__((address_space(1))) v4i* g_v4i_ptr;
typedef __attribute__((address_space(3))) v4i* l_v4i_ptr;

constexpr int BB = 2, SS = 2048, DD = 1024, HH = 16, DK = 64;
constexpr int SCP = SS + 8;                        // LDS row pitch (floats), keeps 16B align
constexpr int SMEM_FLOATS = 16 * SCP + 256 + 32;   // scores + reduction scratch
constexpr size_t SMEM_BYTES = (size_t)SMEM_FLOATS * 4;

// ---------------------------------------------------------------------------
// Fragment loaders per CDNA5 WMMA VGPR layouts (CDNA5 ISA §7.12.2).
// A 16x32 f16 tile: lane 0-15 -> rows M=0..15, K window 0..7 & 16..23;
// lanes 16-31 -> same rows, K window 8..15 & 24..31.  Two b128 loads/lane.
// ---------------------------------------------------------------------------
static __device__ __forceinline__ v16h frag_f16(const _Float16* __restrict__ base,
                                                int ld, int row0, int k0, int lane) {
  const int hf = lane >> 4, r = lane & 15;
  const _Float16* p = base + (size_t)(row0 + r) * ld + (k0 + 8 * hf);
  v8h lo = *(const v8h*)p;
  v8h hi = *(const v8h*)(p + 16);
  return __builtin_shufflevector(lo, hi, 0, 1, 2, 3, 4, 5, 6, 7,
                                 8, 9, 10, 11, 12, 13, 14, 15);
}

static __device__ __forceinline__ v16h cvt16(float4 q0, float4 q1, float4 q2, float4 q3) {
  v16h a;
  a[0]  = (_Float16)q0.x; a[1]  = (_Float16)q0.y; a[2]  = (_Float16)q0.z; a[3]  = (_Float16)q0.w;
  a[4]  = (_Float16)q1.x; a[5]  = (_Float16)q1.y; a[6]  = (_Float16)q1.z; a[7]  = (_Float16)q1.w;
  a[8]  = (_Float16)q2.x; a[9]  = (_Float16)q2.y; a[10] = (_Float16)q2.z; a[11] = (_Float16)q2.w;
  a[12] = (_Float16)q3.x; a[13] = (_Float16)q3.y; a[14] = (_Float16)q3.z; a[15] = (_Float16)q3.w;
  return a;
}

// A-fragment from fp32 scores resident in LDS (rows = 16-row query block).
static __device__ __forceinline__ v16h frag_lds_f32(const float* sc, int k0, int lane) {
  const int hf = lane >> 4, r = lane & 15;
  const float* p = sc + r * SCP + k0 + 8 * hf;
  return cvt16(*(const float4*)(p),      *(const float4*)(p + 4),
               *(const float4*)(p + 16), *(const float4*)(p + 20));
}

// ---------------------------------------------------------------------------
__global__ __launch_bounds__(256) void cvt_f32f16_kernel(const float* __restrict__ in,
                                                         _Float16* __restrict__ out, int n) {
  for (int i = blockIdx.x * blockDim.x + threadIdx.x; i < n; i += gridDim.x * blockDim.x)
    out[i] = (_Float16)in[i];
}

// ---------------------------------------------------------------------------
// C[M,N] = A[M,K] * W[N,K]^T + bias   (torch Linear)
// mode 0: store f16 head layout  [B,H,S,DK]   (Q, K projections)
// mode 1: store f16 transposed   [B,H,DK,S]   (V projection)
// mode 2: store f32 plain [M,N]               (output projection -> d_out)
// ---------------------------------------------------------------------------
__global__ __launch_bounds__(256) void gemm16_kernel(
    const _Float16* __restrict__ A, const _Float16* __restrict__ W,
    const float* __restrict__ bias, _Float16* __restrict__ outh,
    float* __restrict__ outf, int M, int N, int K, int mode) {
  const int wave = threadIdx.x >> 5, lane = threadIdx.x & 31;
  const int tile = blockIdx.x * (blockDim.x >> 5) + wave;
  const int ntn = N >> 4;
  const int mt = tile / ntn, nt = tile - mt * ntn;
  if (mt * 16 >= M) return;

  v8f c = {};
  for (int k0 = 0; k0 < K; k0 += 32) {
    v16h a = frag_f16(A, K, mt * 16, k0, lane);
    v16h b = frag_f16(W, K, nt * 16, k0, lane);
    c = __builtin_amdgcn_wmma_f32_16x16x32_f16(false, a, false, b, (short)0, c, false, false);
  }

  const int hf = lane >> 4, col = lane & 15;
  const int n = nt * 16 + col;
  const float bv = bias[n];
#pragma unroll
  for (int e = 0; e < 8; ++e) {
    const int m = mt * 16 + e + 8 * hf;
    const float v = c[e] + bv;
    if (mode == 2) {
      outf[(size_t)m * N + n] = v;
    } else {
      const int b_ = m >> 11, s = m & (SS - 1);
      const int h = n >> 6, d = n & (DK - 1);
      if (mode == 0)
        outh[((size_t)(b_ * HH + h) * SS + s) * DK + d] = (_Float16)v;
      else
        outh[((size_t)(b_ * HH + h) * DK + d) * SS + s] = (_Float16)v;
    }
  }
}

// ---------------------------------------------------------------------------
// Fused attention: one block per (b, h, 16-query block).
//  1) 8 waves compute causal score tiles (WMMA) into a 16 x S fp32 LDS stripe
//  2) block softmax in LDS
//  3) attn stripe written to HBM once (async store from LDS when available)
//  4) waves 0-3 compute context = P*V (A-frags from LDS, B-frags from Vt)
// ---------------------------------------------------------------------------
__global__ __launch_bounds__(256) void fused_attn_kernel(
    const _Float16* __restrict__ Qh, const _Float16* __restrict__ Kh,
    const _Float16* __restrict__ Vt, float* __restrict__ attn,
    _Float16* __restrict__ ctx) {
  extern __shared__ float smem[];
  float* sc   = smem;                 // [16][SCP] fp32 scores / probabilities
  float* red  = smem + 16 * SCP;      // [256] partial reductions
  float* rmax = red + 256;            // [16] row max
  float* rinv = rmax + 16;            // [16] 1/rowsum

  const int tid = threadIdx.x;
  const int wave = tid >> 5, lane = tid & 31;
  const int hf = lane >> 4, col = lane & 15;

  const int qt = blockIdx.x & 127;
  const int h  = (blockIdx.x >> 7) & 15;
  const int b  = blockIdx.x >> 11;
  const size_t head = (size_t)(b * HH + h);
  const int L = (qt + 1) * 16;        // causal length for this query block

  const _Float16* Qb = Qh + head * SS * DK;
  const _Float16* Kb = Kh + head * SS * DK;

  // ---- Phase 1: score tiles into LDS (kt <= qt), plus zero-fill k >= L ----
  if (wave <= qt) {
    const v16h a0 = frag_f16(Qb, DK, qt * 16, 0, lane);
    const v16h a1 = frag_f16(Qb, DK, qt * 16, 32, lane);
    for (int kt = wave; kt <= qt; kt += 8) {
      v16h b0 = frag_f16(Kb, DK, kt * 16, 0, lane);
      v16h b1 = frag_f16(Kb, DK, kt * 16, 32, lane);
      v8f c = {};
      c = __builtin_amdgcn_wmma_f32_16x16x32_f16(false, a0, false, b0, (short)0, c, false, false);
      c = __builtin_amdgcn_wmma_f32_16x16x32_f16(false, a1, false, b1, (short)0, c, false, false);
      const int k = kt * 16 + col;
#pragma unroll
      for (int e = 0; e < 8; ++e) {
        const int m = e + 8 * hf;
        const int q = qt * 16 + m;
        float v = c[e] * 0.125f;      // 1/sqrt(DK)
        if (k > q) v = -1e9f;         // diagonal tile causal mask
        sc[m * SCP + kt * 16 + col] = v;
      }
    }
  }
  // zero-fill the region beyond the causal length (final attn value there is 0)
  for (int r = 0; r < 16; ++r)
    for (int k = L + 4 * tid; k < SS; k += 1024)
      *(float4*)(sc + r * SCP + k) = make_float4(0.f, 0.f, 0.f, 0.f);
  __syncthreads();

  // ---- Phase 2: softmax over k in [0, L), 16 threads per row ----
  const int r = tid >> 4, cc = tid & 15;
  float mx = -1e30f;
  for (int k = cc * 4; k < L; k += 64) {
    float4 v = *(const float4*)(sc + r * SCP + k);
    mx = fmaxf(mx, fmaxf(fmaxf(v.x, v.y), fmaxf(v.z, v.w)));
  }
  red[tid] = mx;
  __syncthreads();
  if (cc == 0) {
    float m2 = -1e30f;
    for (int j = 0; j < 16; ++j) m2 = fmaxf(m2, red[(r << 4) + j]);
    rmax[r] = m2;
  }
  __syncthreads();
  mx = rmax[r];
  float sum = 0.f;
  for (int k = cc * 4; k < L; k += 64) {
    float4 v = *(float4*)(sc + r * SCP + k);
    v.x = __expf(v.x - mx); v.y = __expf(v.y - mx);
    v.z = __expf(v.z - mx); v.w = __expf(v.w - mx);
    sum += v.x + v.y + v.z + v.w;
    *(float4*)(sc + r * SCP + k) = v;
  }
  red[tid] = sum;
  __syncthreads();
  if (cc == 0) {
    float s2 = 0.f;
    for (int j = 0; j < 16; ++j) s2 += red[(r << 4) + j];
    rinv[r] = 1.0f / s2;
  }
  __syncthreads();
  const float inv = rinv[r];
  for (int k = cc * 4; k < L; k += 64) {
    float4 v = *(float4*)(sc + r * SCP + k);
    v.x *= inv; v.y *= inv; v.z *= inv; v.w *= inv;
    *(float4*)(sc + r * SCP + k) = v;
  }
  __syncthreads();

  // ---- Phase 3: single write of the 16 x S attn stripe to HBM ----
  {
    float* arow = attn + head * SS * SS + (size_t)(qt * 16) * SS;
#pragma unroll 4
    for (int it = 0; it < 32; ++it) {
      const int chunk = tid + 256 * it;           // 8192 16B chunks
      const int row = chunk >> 9;
      const int kk = (chunk & 511) << 2;
      float* gp = arow + (size_t)row * SS + kk;
      const float* lp = sc + row * SCP + kk;
#if __has_builtin(__builtin_amdgcn_global_store_async_from_lds_b128)
      __builtin_amdgcn_global_store_async_from_lds_b128(
          (g_v4i_ptr)(__attribute__((address_space(1))) void*)gp,
          (l_v4i_ptr)(__attribute__((address_space(3))) void*)lp, 0, 0);
#else
      *(float4*)gp = *(const float4*)lp;
#endif
    }
  }

  // ---- Phase 4: context = P * V  (waves 0-3, one 16x16 d-tile each) ----
  if (wave < 4) {
    const int dt = wave;
    const _Float16* Bb = Vt + (head * DK + dt * 16) * SS;
    v8f c = {};
    const int nk = (qt + 2) >> 1;                  // ceil(L/32)
    for (int i = 0; i < nk; ++i) {
      const int k0 = i * 32;
      v16h a = frag_lds_f32(sc, k0, lane);
      v16h bb = frag_f16(Bb, SS, 0, k0, lane);
      c = __builtin_amdgcn_wmma_f32_16x16x32_f16(false, a, false, bb, (short)0, c, false, false);
    }
    const int d = h * DK + dt * 16 + col;
#pragma unroll
    for (int e = 0; e < 8; ++e) {
      const int q = qt * 16 + e + 8 * hf;
      ctx[((size_t)b * SS + q) * DD + d] = (_Float16)c[e];
    }
  }

#if __has_builtin(__builtin_amdgcn_global_store_async_from_lds_b128)
#if __has_builtin(__builtin_amdgcn_s_wait_asynccnt)
  __builtin_amdgcn_s_wait_asynccnt(0);
#else
  asm volatile("s_wait_asynccnt 0x0" ::: "memory");
#endif
#endif
}

// ---------------------------------------------------------------------------
extern "C" void kernel_launch(void* const* d_in, const int* in_sizes, int n_in,
                              void* d_out, int out_size, void* d_ws, size_t ws_size,
                              hipStream_t stream) {
  (void)in_sizes; (void)n_in; (void)out_size; (void)ws_size;
  const float* query = (const float*)d_in[0];
  const float* key_  = (const float*)d_in[1];
  const float* value = (const float*)d_in[2];
  // d_in[3] = causal mask; applied analytically.
  const float* Wq = (const float*)d_in[4];  const float* bq = (const float*)d_in[5];
  const float* Wk = (const float*)d_in[6];  const float* bk = (const float*)d_in[7];
  const float* Wv = (const float*)d_in[8];  const float* bv = (const float*)d_in[9];
  const float* Wo = (const float*)d_in[10]; const float* bo = (const float*)d_in[11];

  float* out  = (float*)d_out;                       // [B,S,D]
  float* attn = out + (size_t)BB * SS * DD;          // [B,H,S,S]

  const size_t M  = (size_t)BB * SS;   // 4096
  const size_t XN = M * DD;            // 4,194,304
  const size_t WN = (size_t)DD * DD;   // 1,048,576

  char* w = (char*)d_ws;
  size_t off = 0;
  auto carve = [&](size_t elems) {
    void* p = w + off;
    off += ((elems * sizeof(_Float16)) + 255) & ~(size_t)255;
    return (_Float16*)p;
  };
  _Float16* Xq16 = carve(XN); _Float16* Xk16 = carve(XN); _Float16* Xv16 = carve(XN);
  _Float16* Wq16 = carve(WN); _Float16* Wk16 = carve(WN);
  _Float16* Wv16 = carve(WN); _Float16* Wo16 = carve(WN);
  _Float16* Qh = carve(XN);   _Float16* Kh = carve(XN);
  _Float16* Vt = carve(XN);   _Float16* Ctx = carve(XN);

  // 1) f32 -> f16 conversions
  cvt_f32f16_kernel<<<2048, 256, 0, stream>>>(query, Xq16, (int)XN);
  cvt_f32f16_kernel<<<2048, 256, 0, stream>>>(key_,  Xk16, (int)XN);
  cvt_f32f16_kernel<<<2048, 256, 0, stream>>>(value, Xv16, (int)XN);
  cvt_f32f16_kernel<<<1024, 256, 0, stream>>>(Wq, Wq16, (int)WN);
  cvt_f32f16_kernel<<<1024, 256, 0, stream>>>(Wk, Wk16, (int)WN);
  cvt_f32f16_kernel<<<1024, 256, 0, stream>>>(Wv, Wv16, (int)WN);
  cvt_f32f16_kernel<<<1024, 256, 0, stream>>>(Wo, Wo16, (int)WN);

  // 2) projections
  const int gemmBlocks = (int)((M / 16) * (DD / 16) / 8);
  gemm16_kernel<<<gemmBlocks, 256, 0, stream>>>(Xq16, Wq16, bq, Qh, nullptr,
                                                (int)M, DD, DD, 0);
  gemm16_kernel<<<gemmBlocks, 256, 0, stream>>>(Xk16, Wk16, bk, Kh, nullptr,
                                                (int)M, DD, DD, 0);
  gemm16_kernel<<<gemmBlocks, 256, 0, stream>>>(Xv16, Wv16, bv, Vt, nullptr,
                                                (int)M, DD, DD, 1);

  // 3) fused scores+softmax+attn-writeback+P*V: B*H*(S/16) = 4096 blocks
  (void)hipFuncSetAttribute((const void*)fused_attn_kernel,
                            hipFuncAttributeMaxDynamicSharedMemorySize,
                            (int)SMEM_BYTES);
  fused_attn_kernel<<<BB * HH * (SS / 16), 256, SMEM_BYTES, stream>>>(Qh, Kh, Vt, attn, Ctx);

  // 4) output projection -> fp32 d_out
  gemm16_kernel<<<gemmBlocks, 256, 0, stream>>>(Ctx, Wo16, bo, nullptr, out,
                                                (int)M, DD, DD, 2);
}
